// HybridNet_40690520162745
// MI455X (gfx1250) — compile-verified, matrix operand order checked by reading, not run on hardware
//
#include <hip/hip_runtime.h>
#include <cstdint>
#include <cstddef>

// ---------------------------------------------------------------------------
// MI455X (gfx1250) implementation.
//  - Conv layers = implicit GEMM via v_wmma_f32_16x16x32_bf16; A operand is
//    im2col-staged in LDS, B staged transposed, so each WMMA fragment half is
//    one aligned ds_load_b128 (no per-element branches/divides).
//  - Activations ping-pong through d_ws as bf16 (pipeline is HBM-bound:
//    ~400MB traffic @ 23.3TB/s >> 24.5 GFLOP of WMMA math).
//  - conv + bias + ReLU + maxpool2 fused per layer.
//  - seq/W1 padded+transposed so the 256x128x1040 projection GEMM uses
//    branch-free global_load_b128 fragments.
//  - LSTM tail is latency-bound: one persistent 256-thread block.
// ---------------------------------------------------------------------------

typedef __attribute__((ext_vector_type(16))) __bf16          v16bf;
typedef __attribute__((ext_vector_type(8)))  unsigned int    v8u;
typedef __attribute__((ext_vector_type(4)))  unsigned int    v4u;
typedef __attribute__((ext_vector_type(8)))  float           v8f;

__device__ __forceinline__ unsigned short f2bf(float x) {
    union { float f; unsigned u; } v; v.f = x;
    unsigned r = v.u + 0x7FFFu + ((v.u >> 16) & 1u);   // round-to-nearest-even
    return (unsigned short)(r >> 16);
}
__device__ __forceinline__ float bf2f(unsigned short x) {
    union { unsigned u; float f; } v; v.u = ((unsigned)x) << 16;
    return v.f;
}

// ------------------------------ f32 -> bf16 --------------------------------
__global__ void cvt_f32_bf16(const float* __restrict__ src,
                             unsigned short* __restrict__ dst, int n) {
    int i = blockIdx.x * blockDim.x + threadIdx.x;
    if (i < n) dst[i] = f2bf(src[i]);
}

// ---- lstm1.W (1040x128) -> bf16 transposed+padded [128][1056] -------------
__global__ void cvt_w1_transpose(const float* __restrict__ src,
                                 unsigned short* __restrict__ dst) {
    int i = blockIdx.x * blockDim.x + threadIdx.x;
    if (i >= 128 * 1056) return;
    int j = i / 1056, k = i - j * 1056;
    dst[i] = (k < 1040) ? f2bf(src[k * 128 + j]) : (unsigned short)0;
}

// ------------------- grouped conv (K=3,SAME) + ReLU + pool -----------------
// One wave per block computes a 16(conv-pos) x 16(out-ch) WMMA tile of one
// (n, group) pair. A is im2col'ed into LDS as sA[m][kk], B transposed into
// sB[col][kk], both zero-padded to Kpad (multiple of 32) so the K loop is
// branch-free: fragment halves are aligned 128-bit LDS loads.
// Activation layout: [n][c][L][ch] (bf16); weights [kk=k*in_ch+ic][8*f] (bf16)
__global__ __launch_bounds__(32)
void conv_relu_pool(const unsigned short* __restrict__ actIn,
                    const unsigned short* __restrict__ wpack,
                    const float* __restrict__ bias,
                    unsigned short* __restrict__ actOut,
                    int L_in, int in_ch, int log2in, int f)
{
    __shared__ __align__(16) unsigned short sA[16 * 384];  // 12KB max
    __shared__ __align__(16) unsigned short sB[16 * 384];  // 12KB max

    const int tid   = threadIdx.x;
    const int lane  = tid & 15;
    const int half  = tid >> 4;
    const int hbase = half * 8;

    const int nc       = blockIdx.z;            // n*8 + c
    const int c        = nc & 7;
    const int pos_base = blockIdx.x * 16;
    const int oc_base  = blockIdx.y * 16;
    const int Ktot     = 3 * in_ch;
    const int Kpad     = (Ktot + 31) & ~31;
    const int L_out    = L_in >> 1;
    const int CF       = 8 * f;
    const size_t wcol  = (size_t)c * f + oc_base;

    __builtin_prefetch(wpack + wcol, 0, 1);     // global_prefetch_b8

    // --- stage A: im2col, zero-padded ---
    const size_t inBase = (size_t)nc * L_in;
    for (int m = 0; m < 16; ++m) {
        for (int kk = tid; kk < Kpad; kk += 32) {
            int k  = kk >> log2in;              // tap 0..2 (garbage if padded)
            int ic = kk & (in_ch - 1);
            int p  = pos_base + m + k - 1;      // SAME pad, K=3
            unsigned short v = 0;
            if (kk < Ktot && p >= 0 && p < L_in)
                v = actIn[(inBase + p) * in_ch + ic];
            sA[m * Kpad + kk] = v;
        }
    }
    // --- stage B transposed: sB[col][kk], zero-padded tail ---
    for (int i = tid; i < Ktot * 16; i += 32) {
        int kk = i >> 4, col = i & 15;
        sB[col * Kpad + kk] = wpack[(size_t)kk * CF + wcol + col];
    }
    for (int i = Ktot * 16 + tid; i < Kpad * 16; i += 32) {
        int kk = i >> 4, col = i & 15;
        sB[col * Kpad + kk] = 0;
    }
    __syncthreads();

    // --- branch-free K loop: 4x ds_load_b128 + 1 wmma per 32-K chunk ---
    // Fragment element mapping (16-bit A 16x32 / B 32x16):
    //   elems 0-7  <-> K = kb + hbase + 0..7   (contiguous)
    //   elems 8-15 <-> K = kb + 16 + hbase + 0..7
    v8f acc = {};
    const v4u* pA = (const v4u*)sA;             // 8 bf16 per v4u
    const v4u* pB = (const v4u*)sB;
    const int fo = (lane * Kpad + hbase) >> 3;  // row/col fragment base
    const int nq = Kpad >> 3;                   // v4u per row
    for (int q = 0; q < nq; q += 4) {
        v4u a0 = pA[fo + q];
        v4u a1 = pA[fo + q + 2];
        v4u b0 = pB[fo + q];
        v4u b1 = pB[fo + q + 2];
        v8u a8 = __builtin_shufflevector(a0, a1, 0, 1, 2, 3, 4, 5, 6, 7);
        v8u b8 = __builtin_shufflevector(b0, b1, 0, 1, 2, 3, 4, 5, 6, 7);
        acc = __builtin_amdgcn_wmma_f32_16x16x32_bf16(
                  false, __builtin_bit_cast(v16bf, a8),
                  false, __builtin_bit_cast(v16bf, b8),
                  (short)0, acc, false, false);
    }

    // --- epilogue: bias + ReLU + maxpool2 out of the C fragment ---
    // C/D layout: VGPR r holds M = r + half*8, column N = lane; pool pairs
    // (2m,2m+1) are adjacent VGPRs in the same lane.
    const float bf = bias[wcol + lane];
    const int l2base = (pos_base >> 1) + half * 4;
#pragma unroll
    for (int r = 0; r < 8; r += 2) {
        float v0 = fmaxf(acc[r]     + bf, 0.f);
        float v1 = fmaxf(acc[r + 1] + bf, 0.f);
        float pv = fmaxf(v0, v1);
        int lp = l2base + (r >> 1);
        if (lp < L_out)
            actOut[((size_t)nc * L_out + lp) * f + oc_base + lane] = f2bf(pv);
    }
}

// -------------------- BN1 + concat(input_2) -> seq (bf16) ------------------
// seq is row-padded to 1056 (zero tail) so the GEMM K loop needs no guards.
__global__ void build_seq(const unsigned short* __restrict__ act,  // [n][1024]
                          const float* __restrict__ in2,           // [n][16]
                          const float* __restrict__ g,  const float* __restrict__ bta,
                          const float* __restrict__ mean, const float* __restrict__ var,
                          unsigned short* __restrict__ seq)        // [n][1056]
{
    int i = blockIdx.x * blockDim.x + threadIdx.x;
    if (i >= 256 * 1056) return;
    int n = i / 1056, k = i - n * 1056;
    float v = 0.f;
    if (k < 1024) {
        float x = bf2f(act[n * 1024 + k]);
        v = g[k] * (x - mean[k]) * rsqrtf(var[k] + 1e-3f) + bta[k];
    } else if (k < 1040) {
        v = in2[n * 16 + (k - 1024)];
    }
    seq[i] = f2bf(v);
}

// ----------------- xz = seq(256x1040) @ W(1040x128) + b --------------------
// A rows and (transposed) B rows are contiguous in global memory: fragment
// halves are direct aligned global_load_b128, branch-free over 33 K-chunks.
__global__ __launch_bounds__(32)
void gemm_xz(const unsigned short* __restrict__ seq,   // [256][1056] bf16
             const unsigned short* __restrict__ wt,    // [128][1056] bf16 (W^T)
             const float* __restrict__ b,
             float* __restrict__ xz)                   // [256][128]
{
    const int tid = threadIdx.x, lane = tid & 15, half = tid >> 4, hbase = half * 8;
    const int m_base = blockIdx.x * 16;
    const int j_base = blockIdx.y * 16;
    const int Kp = 1056;

    const v4u* pA = (const v4u*)(seq + (size_t)(m_base + lane) * Kp);
    const v4u* pB = (const v4u*)(wt  + (size_t)(j_base + lane) * Kp);
    const int fo = hbase >> 3;                  // 0 or 1
    v8f acc = {};
    for (int q = 0; q < (Kp >> 3); q += 4) {    // 33 chunks of K=32
        v4u a0 = pA[fo + q];
        v4u a1 = pA[fo + q + 2];
        v4u b0 = pB[fo + q];
        v4u b1 = pB[fo + q + 2];
        v8u a8 = __builtin_shufflevector(a0, a1, 0, 1, 2, 3, 4, 5, 6, 7);
        v8u b8 = __builtin_shufflevector(b0, b1, 0, 1, 2, 3, 4, 5, 6, 7);
        acc = __builtin_amdgcn_wmma_f32_16x16x32_bf16(
                  false, __builtin_bit_cast(v16bf, a8),
                  false, __builtin_bit_cast(v16bf, b8),
                  (short)0, acc, false, false);
    }
    const float bj = b[j_base + lane];
#pragma unroll
    for (int r = 0; r < 8; ++r) {
        int n = m_base + r + hbase;             // C/D: M = r + half*8
        xz[(size_t)n * 128 + j_base + lane] = acc[r] + bj;
    }
}

// ----------- both LSTMs (sequential) + BN2 + dense chain + sigmoid ---------
__global__ __launch_bounds__(256)
void lstm_dense(const float* __restrict__ xz,       // [b][t][128] (bias folded)
                const float* __restrict__ U1,       // [32][128]
                const float* __restrict__ W2, const float* __restrict__ U2,
                const float* __restrict__ b2,
                const float* __restrict__ g2, const float* __restrict__ bt2,
                const float* __restrict__ m2, const float* __restrict__ v2,
                const float* __restrict__ dw1, const float* __restrict__ db1,
                const float* __restrict__ dw2, const float* __restrict__ db2,
                const float* __restrict__ dw3, const float* __restrict__ db3,
                const float* __restrict__ dw4, const float* __restrict__ db4,
                float* __restrict__ out)
{
    __shared__ float h1[256], c1[256], h2[256], c2[256];
    __shared__ float zb[1024];
    __shared__ float ab[2048];
    const int tid = threadIdx.x;
    const int b = tid >> 5, jj = tid & 31;          // (batch, hidden unit)
    h1[tid] = 0.f; c1[tid] = 0.f; h2[tid] = 0.f; c2[tid] = 0.f;
    __syncthreads();

    for (int t = 0; t < 32; ++t) {
        // z1 = xz[:,t,:] + h1 @ U1
        for (int i = tid; i < 1024; i += 256) {
            int bb = i >> 7, j = i & 127;
            float s = xz[(size_t)(bb * 32 + t) * 128 + j];
            for (int k = 0; k < 32; ++k) s += h1[bb * 32 + k] * U1[k * 128 + j];
            zb[i] = s;
        }
        __syncthreads();
        {   // gates i,f,g,o ; cell activation = relu (per reference)
            float zi = zb[b * 128 + jj],      zf = zb[b * 128 + 32 + jj];
            float zg = zb[b * 128 + 64 + jj], zo = zb[b * 128 + 96 + jj];
            float ig = 1.f / (1.f + expf(-zi));
            float fg = 1.f / (1.f + expf(-zf));
            float og = 1.f / (1.f + expf(-zo));
            float cn = fg * c1[tid] + ig * fmaxf(zg, 0.f);
            c1[tid] = cn;
            h1[tid] = og * fmaxf(cn, 0.f);
        }
        __syncthreads();
        // z2 = b2 + h1 @ W2 + h2 @ U2
        for (int i = tid; i < 1024; i += 256) {
            int bb = i >> 7, j = i & 127;
            float s = b2[j];
            for (int k = 0; k < 32; ++k)
                s += h1[bb * 32 + k] * W2[k * 128 + j]
                   + h2[bb * 32 + k] * U2[k * 128 + j];
            zb[i] = s;
        }
        __syncthreads();
        {
            float zi = zb[b * 128 + jj],      zf = zb[b * 128 + 32 + jj];
            float zg = zb[b * 128 + 64 + jj], zo = zb[b * 128 + 96 + jj];
            float ig = 1.f / (1.f + expf(-zi));
            float fg = 1.f / (1.f + expf(-zf));
            float og = 1.f / (1.f + expf(-zo));
            float cn = fg * c2[tid] + ig * fmaxf(zg, 0.f);
            c2[tid] = cn;
            h2[tid] = og * fmaxf(cn, 0.f);
        }
        __syncthreads();
    }

    // BN2
    zb[tid] = g2[jj] * (h2[tid] - m2[jj]) * rsqrtf(v2[jj] + 1e-3f) + bt2[jj];
    __syncthreads();
    // dense1: 32 -> 256 (relu)
    for (int i = tid; i < 2048; i += 256) {
        int bb = i >> 8, o = i & 255;
        float s = db1[o];
        for (int k = 0; k < 32; ++k) s += zb[bb * 32 + k] * dw1[k * 256 + o];
        ab[i] = fmaxf(s, 0.f);
    }
    __syncthreads();
    // dense2: 256 -> 64 (relu)
    for (int i = tid; i < 512; i += 256) {
        int bb = i >> 6, o = i & 63;
        float s = db2[o];
        for (int k = 0; k < 256; ++k) s += ab[bb * 256 + k] * dw2[k * 64 + o];
        zb[i] = fmaxf(s, 0.f);
    }
    __syncthreads();
    // dense3: 64 -> 16 (relu)
    if (tid < 128) {
        int bb = tid >> 4, o = tid & 15;
        float s = db3[o];
        for (int k = 0; k < 64; ++k) s += zb[bb * 64 + k] * dw3[k * 16 + o];
        ab[tid] = fmaxf(s, 0.f);
    }
    __syncthreads();
    // dense4: 16 -> 1 (sigmoid)
    if (tid < 8) {
        float s = db4[0];
        for (int k = 0; k < 16; ++k) s += ab[tid * 16 + k] * dw4[k];
        out[tid] = 1.f / (1.f + expf(-s));
    }
}

// ---------------------------------------------------------------------------
extern "C" void kernel_launch(void* const* d_in, const int* in_sizes, int n_in,
                              void* d_out, int out_size, void* d_ws, size_t ws_size,
                              hipStream_t stream)
{
    (void)in_sizes; (void)n_in; (void)out_size; (void)ws_size;

    // ---- input pointers (setup_inputs dict order) ----
    const float* input_1 = (const float*)d_in[0];
    const float* input_2 = (const float*)d_in[1];
    const float* conv_w[12]; const float* conv_b[12];
    for (int i = 0; i < 12; ++i) {
        conv_w[i] = (const float*)d_in[2 + i];
        conv_b[i] = (const float*)d_in[14 + i];
    }
    const float* bn1g = (const float*)d_in[26];
    const float* bn1b = (const float*)d_in[27];
    const float* bn1m = (const float*)d_in[28];
    const float* bn1v = (const float*)d_in[29];
    const float* bn2g = (const float*)d_in[30];
    const float* bn2b = (const float*)d_in[31];
    const float* bn2m = (const float*)d_in[32];
    const float* bn2v = (const float*)d_in[33];
    const float* l1W = (const float*)d_in[34];
    const float* l1U = (const float*)d_in[35];
    const float* l1b = (const float*)d_in[36];
    const float* l2W = (const float*)d_in[37];
    const float* l2U = (const float*)d_in[38];
    const float* l2b = (const float*)d_in[39];
    const float* dw1 = (const float*)d_in[40]; const float* db1 = (const float*)d_in[41];
    const float* dw2 = (const float*)d_in[42]; const float* db2 = (const float*)d_in[43];
    const float* dw3 = (const float*)d_in[44]; const float* db3 = (const float*)d_in[45];
    const float* dw4 = (const float*)d_in[46]; const float* db4 = (const float*)d_in[47];
    float* out = (float*)d_out;

    // ---- layer tables ----
    static const int Lin[12] = {4096,2048,1024,512,256,128,64,32,16,8,4,2};
    static const int inc[12] = {1,16,16,16,32,32,32,64,64,64,128,128};
    static const int lg2[12] = {0,4,4,4,5,5,5,6,6,6,7,7};
    static const int ff [12] = {16,16,16,32,32,32,64,64,64,128,128,128};

    // ---- workspace carve-up (256B aligned) ----
    char* base = (char*)d_ws;
    size_t off = 0;
    auto take = [&](size_t bytes) -> char* {
        char* p = base + off;
        off = (off + bytes + 255) & ~(size_t)255;
        return p;
    };
    unsigned short* wbf[12];
    int welems[12];
    for (int i = 0; i < 12; ++i) {
        welems[i] = 3 * inc[i] * 8 * ff[i];
        wbf[i] = (unsigned short*)take((size_t)welems[i] * 2);
    }
    unsigned short* w1t  = (unsigned short*)take((size_t)128 * 1056 * 2);
    unsigned short* seq  = (unsigned short*)take((size_t)256 * 1056 * 2);
    float*          xz   = (float*)         take((size_t)256 * 128 * 4);
    unsigned short* buf0 = (unsigned short*)take((size_t)33554432 * 2); // <=67MB
    unsigned short* buf1 = (unsigned short*)take((size_t)67108864 * 2); // <=134MB

    // ---- pack input + all weights to bf16 ----
    {
        int n = 8 * 32 * 8 * 4096;   // input_1 elements, layout already [n][c][l]
        cvt_f32_bf16<<<(n + 255) / 256, 256, 0, stream>>>(input_1, buf0, n);
    }
    for (int i = 0; i < 12; ++i)
        cvt_f32_bf16<<<(welems[i] + 255) / 256, 256, 0, stream>>>(conv_w[i], wbf[i], welems[i]);
    {
        int n = 128 * 1056;          // W1 -> transposed + padded bf16
        cvt_w1_transpose<<<(n + 255) / 256, 256, 0, stream>>>(l1W, w1t);
    }

    // ---- 12 fused conv+relu+pool layers, ping-pong bf16 activations ----
    unsigned short* cin = buf0;
    for (int i = 0; i < 12; ++i) {
        unsigned short* cout = (i % 2 == 0) ? buf1 : buf0;
        dim3 grid((Lin[i] + 15) / 16, ff[i] / 16, 8 * 32 * 8); // pos, oc, n*c
        conv_relu_pool<<<grid, 32, 0, stream>>>(cin, wbf[i], conv_b[i], cout,
                                                Lin[i], inc[i], lg2[i], ff[i]);
        cin = cout;
    }
    // after layer 12 (i=11, odd) the result sits in buf0; cin == buf0

    // ---- BN1 + concat -> seq (padded) ----
    {
        int n = 256 * 1056;
        build_seq<<<(n + 255) / 256, 256, 0, stream>>>(cin, input_2,
                                                       bn1g, bn1b, bn1m, bn1v, seq);
    }
    // ---- input projection GEMM (bias fused) ----
    gemm_xz<<<dim3(16, 8), 32, 0, stream>>>(seq, w1t, l1b, xz);

    // ---- sequential LSTMs + BN2 + dense chain ----
    lstm_dense<<<1, 256, 0, stream>>>(xz, l1U, l2W, l2U, l2b,
                                      bn2g, bn2b, bn2m, bn2v,
                                      dw1, db1, dw2, db2, dw3, db3, dw4, db4,
                                      out);
}
